// SimpleRGAT_25391846654703
// MI455X (gfx1250) — matrix operand order repaired and checked by reading, compiled, not run on hardware
//
#include <hip/hip_runtime.h>

// ---------------------------------------------------------------------------
// SimpleRGAT for MI455X (gfx1250, wave32, WMMA) -- round 3
//
//   K0: convert Wq/Wk/Wv to fp16 (weights stay L2-resident)
//   K1: zero red[N,128], den[N,4]
//   K2: q = h @ Wq^T     (8 waves/block, wave owns one N-tile, B-frags held
//                         in registers across 8 row-tiles; full blocks take a
//                         fully-unrolled path; stores use base+imm offsets)
//   K3: fused edge kernel, 128 edges per block:
//         gather hs = leaky_relu(h[src]+relvec[id]) -> LDS (f16)
//         k = hs@Wk^T  (per-wave N-tile, B resident)   -> LDS (f32)
//         logits/exp -> exS, atomicAdd den
//         v = hs@Wv^T recomputed in-register, atomicAdd red straight from
//         the WMMA C-fragment layout (v never touches LDS)
//   K4: out = h + celu(red/den)
// ---------------------------------------------------------------------------

#define N_NODES 50000
#define N_EDGES 600000
#define HDIM    128
#define N_HEADS 4
#define DH      32
#define INV_SQRT_DH 0.17677669529663687f

#define QTILES (N_NODES / 16)   // 3125
#define ETILES (N_EDGES / 16)   // 37500
#define TB     8                // row/edge tiles per 256-thread block

typedef __attribute__((ext_vector_type(16))) _Float16 v16h;
typedef __attribute__((ext_vector_type(8)))  float    v8f;

// ---- WMMA fragment helpers (wave32 layouts per CDNA5 ISA 7.12.2) -----------

// A 16x32 f16: lanes 0-15 -> M=lane, K group base 0; lanes 16-31 -> M=lane-16,
// K group base +8.  Half e: K = kb + e (+8 for e>=8).
__device__ __forceinline__ v16h load_a_frag(const _Float16* __restrict__ smem,
                                            int lane, int kc) {
  const int m  = lane & 15;
  const int kb = kc * 32 + ((lane >> 4) << 3);
  v16h a;
#pragma unroll
  for (int e = 0; e < 16; ++e) {
    const int k = kb + e + ((e >= 8) ? 8 : 0);
    a[e] = smem[m * HDIM + k];
  }
  return a;
}

// B 32x16 f16 (B[k][n] = W[n][k], W row-major): lane holds column n=lane&15;
// same K order as A: K = kc*32 + 8*(lane>>4) + e (+8 for e>=8).
__device__ __forceinline__ v16h load_b_frag(const _Float16* __restrict__ W16,
                                            int lane, int nt, int kc) {
  const int n  = nt * 16 + (lane & 15);
  const int kb = kc * 32 + ((lane >> 4) << 3);
  v16h b;
#pragma unroll
  for (int e = 0; e < 16; ++e) b[e] = W16[n * HDIM + kb + e + ((e >= 8) ? 8 : 0)];
  return b;
}

__device__ __forceinline__ v8f wmma_f16(v16h a, v16h b, v8f c) {
  return __builtin_amdgcn_wmma_f32_16x16x32_f16(false, a, false, b,
                                                (short)0, c, false, false);
}

// ---- K0 ---------------------------------------------------------------------
__global__ void rgat_convert_w_kernel(const float* __restrict__ Wq,
                                      const float* __restrict__ Wk,
                                      const float* __restrict__ Wv,
                                      _Float16* __restrict__ Wq16,
                                      _Float16* __restrict__ Wk16,
                                      _Float16* __restrict__ Wv16) {
  const int i = blockIdx.x * blockDim.x + threadIdx.x;
  if (i < HDIM * HDIM) {
    Wq16[i] = (_Float16)Wq[i];
    Wk16[i] = (_Float16)Wk[i];
    Wv16[i] = (_Float16)Wv[i];
  }
}

// ---- K1 ---------------------------------------------------------------------
__global__ void rgat_zero_kernel(float* __restrict__ p, int n) {
  const int i = blockIdx.x * blockDim.x + threadIdx.x;
  if (i < n) p[i] = 0.0f;
}

// ---- K2: q projection -------------------------------------------------------
__global__ void __launch_bounds__(256)
rgat_qproj_kernel(const float* __restrict__ h,
                  const _Float16* __restrict__ Wq16,
                  float* __restrict__ q) {
  __shared__ _Float16 aS[TB * 16 * HDIM];          // 32 KB
  const int tid   = threadIdx.x;
  const int lane  = tid & 31;
  const int wave  = tid >> 5;                      // wave == N-tile it owns
  const int tile0 = blockIdx.x * TB;
  int T = QTILES - tile0; if (T > TB) T = TB;      // block-uniform trip count
  const int rows = T * 16;

  // stage rows tile0*16 .. +rows-1 as f16 (2 threads per row, 64 dims each)
  {
    const int m  = tid >> 1;
    const int d0 = (tid & 1) * 64;
    if (m < rows) {
      const float* src = h + (size_t)(tile0 * 16 + m) * HDIM + d0;
      _Float16* dst = aS + m * HDIM + d0;
#pragma unroll
      for (int j = 0; j < 64; j += 4) {
        const float4 hv = *reinterpret_cast<const float4*>(src + j);
        dst[j + 0] = (_Float16)hv.x;
        dst[j + 1] = (_Float16)hv.y;
        dst[j + 2] = (_Float16)hv.z;
        dst[j + 3] = (_Float16)hv.w;
      }
    }
  }
  __syncthreads();

  // B fragments for this wave's N-tile stay in registers for all T row-tiles
  v16h b[4];
#pragma unroll
  for (int kc = 0; kc < 4; ++kc) b[kc] = load_b_frag(Wq16, lane, wave, kc);

  const int col = wave * 16 + (lane & 15);
  const int rb  = (lane >> 4) << 3;

  auto tile_body = [&](int tt) {
    v8f acc = {};
#pragma unroll
    for (int kc = 0; kc < 4; ++kc)
      acc = wmma_f16(load_a_frag(aS + tt * 16 * HDIM, lane, kc), b[kc], acc);
    // one base pointer; the 8 fragment rows are +r*512B -> immediate offsets
    float* qp = q + ((size_t)(tile0 * 16 + tt * 16 + rb) * HDIM + col);
#pragma unroll
    for (int r = 0; r < 8; ++r) qp[r * HDIM] = acc[r];
  };

  if (T == TB) {
#pragma unroll
    for (int tt = 0; tt < TB; ++tt) tile_body(tt);   // full-block fast path
  } else {
    for (int tt = 0; tt < T; ++tt) tile_body(tt);    // single tail block
  }
}

// ---- K3: fused edge kernel --------------------------------------------------
__global__ void __launch_bounds__(256)
rgat_edge_kernel(const float* __restrict__ h,
                 const float* __restrict__ relvec,
                 const int* __restrict__ esrc,
                 const int* __restrict__ edst,
                 const int* __restrict__ eid,
                 const float* __restrict__ q,
                 const _Float16* __restrict__ Wk16,
                 const _Float16* __restrict__ Wv16,
                 float* __restrict__ den,
                 float* __restrict__ red) {
  __shared__ _Float16 hsS[TB * 16 * HDIM];         // 32 KB message tile
  __shared__ float    kS[TB * 16 * HDIM];          // 64 KB k tile
  __shared__ float    exS[TB * 16 * N_HEADS];      //  2 KB softmax weights
  __shared__ int      srcS[TB * 16], dstS[TB * 16], idS[TB * 16];

  const int tid   = threadIdx.x;
  const int lane  = tid & 31;
  const int wave  = tid >> 5;                      // wave == N-tile it owns
  const int tile0 = blockIdx.x * TB;
  int T = ETILES - tile0; if (T > TB) T = TB;
  const int nEdge = T * 16;
  const int e0    = tile0 * 16;

  if (tid < nEdge) {
    srcS[tid] = esrc[e0 + tid];
    dstS[tid] = edst[e0 + tid];
    idS[tid]  = eid[e0 + tid];
  }
  __syncthreads();

  // gather + message: hs = leaky_relu(h[src] + relvec[rel], 0.25) -> f16 LDS
  {
    const int m  = tid >> 1;
    const int d0 = (tid & 1) * 64;
    if (m < nEdge) {
      const float* hr = h + (size_t)srcS[m] * HDIM + d0;
      const float* rv = relvec + (size_t)idS[m] * HDIM + d0;
      _Float16* dst = hsS + m * HDIM + d0;
#pragma unroll
      for (int j = 0; j < 64; j += 4) {
        const float4 hv = *reinterpret_cast<const float4*>(hr + j);
        const float4 rr = *reinterpret_cast<const float4*>(rv + j);
        const float x0 = hv.x + rr.x, x1 = hv.y + rr.y;
        const float x2 = hv.z + rr.z, x3 = hv.w + rr.w;
        dst[j + 0] = (_Float16)(x0 > 0.f ? x0 : 0.25f * x0);
        dst[j + 1] = (_Float16)(x1 > 0.f ? x1 : 0.25f * x1);
        dst[j + 2] = (_Float16)(x2 > 0.f ? x2 : 0.25f * x2);
        dst[j + 3] = (_Float16)(x3 > 0.f ? x3 : 0.25f * x3);
      }
    }
  }
  __syncthreads();

  const int col = wave * 16 + (lane & 15);
  const int rb  = (lane >> 4) << 3;

  // k = hs @ Wk^T : B-frags resident across all T edge tiles
  {
    v16h bk[4];
#pragma unroll
    for (int kc = 0; kc < 4; ++kc) bk[kc] = load_b_frag(Wk16, lane, wave, kc);

    auto k_body = [&](int tt) {
      v8f acc = {};
#pragma unroll
      for (int kc = 0; kc < 4; ++kc)
        acc = wmma_f16(load_a_frag(hsS + tt * 16 * HDIM, lane, kc), bk[kc], acc);
      float* kp = kS + (tt * 16 + rb) * HDIM + col;   // ds imm offsets r*512B
#pragma unroll
      for (int r = 0; r < 8; ++r) kp[r * HDIM] = acc[r];
    };

    if (T == TB) {
#pragma unroll
      for (int tt = 0; tt < TB; ++tt) k_body(tt);
    } else {
      for (int tt = 0; tt < T; ++tt) k_body(tt);
    }
  }
  __syncthreads();

  // un-normalized softmax: ex = exp(<q[dst],k>/sqrt(Dh)); den += ex
  for (int t = tid; t < nEdge * N_HEADS; t += 256) {
    const int m  = t >> 2;
    const int hd = t & 3;
    const int dst = dstS[m];
    const float* qr = q + (size_t)dst * HDIM + hd * DH;
    const float* kr = kS + m * HDIM + hd * DH;
    float s = 0.f;
#pragma unroll
    for (int j = 0; j < DH; ++j) s = fmaf(qr[j], kr[j], s);
    const float ex = __expf(s * INV_SQRT_DH);
    exS[t] = ex;
    atomicAdd(den + (size_t)dst * N_HEADS + hd, ex);
  }
  __syncthreads();

  // v = hs @ Wv^T recomputed in-register; scatter ex*v straight from the
  // C-fragment layout: element r of the v8f is (row rb+r, this wave's col).
  {
    v16h bv[4];
#pragma unroll
    for (int kc = 0; kc < 4; ++kc) bv[kc] = load_b_frag(Wv16, lane, wave, kc);
    const int hd = wave >> 1;                      // head of this N-tile

    auto v_body = [&](int tt) {
      v8f acc = {};
#pragma unroll
      for (int kc = 0; kc < 4; ++kc)
        acc = wmma_f16(load_a_frag(hsS + tt * 16 * HDIM, lane, kc), bv[kc], acc);
#pragma unroll
      for (int r = 0; r < 8; ++r) {
        const int m = tt * 16 + rb + r;
        atomicAdd(red + (size_t)dstS[m] * HDIM + col,
                  exS[m * N_HEADS + hd] * acc[r]);
      }
    };

    if (T == TB) {
#pragma unroll
      for (int tt = 0; tt < TB; ++tt) v_body(tt);
    } else {
      for (int tt = 0; tt < T; ++tt) v_body(tt);
    }
  }
}

// ---- K4 ---------------------------------------------------------------------
__global__ void rgat_finalize_kernel(const float* __restrict__ h,
                                     const float* __restrict__ red,
                                     const float* __restrict__ den,
                                     float* __restrict__ out, int n) {
  const int i = blockIdx.x * blockDim.x + threadIdx.x;
  if (i >= n) return;
  const int node = i >> 7;
  const int hd   = (i & 127) >> 5;
  const float dn = den[node * N_HEADS + hd];
  const float r  = (dn > 0.f) ? (red[i] / dn) : 0.f;
  const float c  = (r > 0.f) ? r : (__expf(r) - 1.f);   // celu, alpha=1
  out[i] = h[i] + c;
}

// ---------------------------------------------------------------------------
extern "C" void kernel_launch(void* const* d_in, const int* in_sizes, int n_in,
                              void* d_out, int out_size, void* d_ws,
                              size_t ws_size, hipStream_t stream) {
  const float* h      = (const float*)d_in[0];
  const float* Wq     = (const float*)d_in[1];
  const float* Wk     = (const float*)d_in[2];
  const float* Wv     = (const float*)d_in[3];
  const float* relvec = (const float*)d_in[4];
  const int*   esrc   = (const int*)d_in[5];
  const int*   edst   = (const int*)d_in[6];
  const int*   eid    = (const int*)d_in[7];
  float*       out    = (float*)d_out;

  // workspace layout (~52 MB): q | red | den | Wq16 | Wk16 | Wv16
  char* ws = (char*)d_ws;
  float* q   = (float*)ws;  ws += (size_t)N_NODES * HDIM * sizeof(float);
  float* red = (float*)ws;  ws += (size_t)N_NODES * HDIM * sizeof(float);
  float* den = (float*)ws;  ws += (size_t)N_NODES * N_HEADS * sizeof(float);
  _Float16* Wq16 = (_Float16*)ws;  ws += (size_t)HDIM * HDIM * sizeof(_Float16);
  _Float16* Wk16 = (_Float16*)ws;  ws += (size_t)HDIM * HDIM * sizeof(_Float16);
  _Float16* Wv16 = (_Float16*)ws;

  rgat_convert_w_kernel<<<(HDIM * HDIM + 255) / 256, 256, 0, stream>>>(
      Wq, Wk, Wv, Wq16, Wk16, Wv16);

  const int nz = N_NODES * (HDIM + N_HEADS);       // red + den contiguous
  rgat_zero_kernel<<<(nz + 255) / 256, 256, 0, stream>>>(red, nz);

  rgat_qproj_kernel<<<(QTILES + TB - 1) / TB, 256, 0, stream>>>(h, Wq16, q);

  rgat_edge_kernel<<<(ETILES + TB - 1) / TB, 256, 0, stream>>>(
      h, relvec, esrc, edst, eid, q, Wk16, Wv16, den, red);

  rgat_finalize_kernel<<<(N_NODES * HDIM + 255) / 256, 256, 0, stream>>>(
      h, red, den, out, N_NODES * HDIM);
}
